// DANetHead_68839735820810
// MI455X (gfx1250) — compile-verified, matrix operand order checked by reading, not run on hardware
//
#include <hip/hip_runtime.h>
#include <hip/hip_bf16.h>
#include <math.h>

typedef unsigned short u16;
typedef unsigned int   u32;
typedef __attribute__((ext_vector_type(16))) __bf16 v16bf;
typedef __attribute__((ext_vector_type(8)))  float  v8f;

struct U4 { u32 x, y, z, w; };
union Frag { v16bf v; U4 q[2]; };

__device__ __forceinline__ u16 f2bf(float f) {
  u32 u = __float_as_uint(f);
  u32 r = (u + 0x7FFFu + ((u >> 16) & 1u)) >> 16;
  return (u16)r;
}

__device__ __forceinline__ Frag load_frag(const u16* p) {
  Frag f;
  f.q[0] = *reinterpret_cast<const U4*>(p);
  f.q[1] = *reinterpret_cast<const U4*>(p + 16);
  return f;
}

__device__ __forceinline__ v8f v8zero() {
  v8f z;
#pragma unroll
  for (int i = 0; i < 8; ++i) z[i] = 0.f;
  return z;
}

#define WMMA_BF16(A, B, C) \
  __builtin_amdgcn_wmma_f32_16x16x32_bf16(false, (A), false, (B), (short)0, (C), false, false)

// ---------------- workspace layout (bytes) ----------------
static const size_t SZ_XPAD   = (size_t)4 * 66 * 66 * 512 * 2;   // 17,842,176
static const size_t SZ_PAD32  = (size_t)2 * 4 * 66 * 66 * 32 * 2;//  2,230,272
static const size_t SZ_WA     = (size_t)2 * 32 * 4608 * 2;
static const size_t SZ_WB     = (size_t)2 * 32 * 288 * 2;
static const size_t SZ_BIAS   = 256;
static const size_t SZ_FEAT   = (size_t)2 * 4 * 32 * 4096 * 4;
static const size_t SZ_FEATBF = (size_t)2 * 4 * 32 * 4096 * 2;
static const size_t SZ_QK     = (size_t)4 * 4 * 4096 * 4;
static const size_t SZ_VBF    = (size_t)4 * 32 * 4096 * 2;
static const size_t SZ_ATT    = (size_t)4096 * 4096 * 2;
static const size_t SZ_SASC   = (size_t)4 * 32 * 4096 * 4;
static const size_t SZ_CONVO  = (size_t)2 * 4 * 32 * 4096 * 4;

static const size_t OFF_XPAD   = 0;
static const size_t OFF_PAD32  = OFF_XPAD + SZ_XPAD;
static const size_t OFF_WA     = OFF_PAD32 + SZ_PAD32;
static const size_t OFF_WB     = OFF_WA + SZ_WA;
static const size_t OFF_BIASA  = OFF_WB + SZ_WB;
static const size_t OFF_BIASB  = OFF_BIASA + SZ_BIAS;
static const size_t OFF_FEAT   = OFF_BIASB + SZ_BIAS;
static const size_t OFF_FEATBF = OFF_FEAT + SZ_FEAT;
static const size_t OFF_Q      = OFF_FEATBF + SZ_FEATBF;
static const size_t OFF_K      = OFF_Q + SZ_QK;
static const size_t OFF_VBF    = OFF_K + SZ_QK;
static const size_t OFF_ATT    = OFF_VBF + SZ_VBF;
static const size_t OFF_SA     = OFF_ATT + SZ_ATT;
static const size_t OFF_SC     = OFF_SA + SZ_SASC;
static const size_t OFF_CONVO  = OFF_SC + SZ_SASC;

// ---------------- kernels ----------------

__global__ __launch_bounds__(256) void zero_u4(U4* p, long n) {
  long i = (long)blockIdx.x * 256 + threadIdx.x;
  if (i < n) { U4 z; z.x = z.y = z.z = z.w = 0u; p[i] = z; }
}

// fold BN scale into 512->32 conv weights; layout [br][o][tap*512 + c]
__global__ __launch_bounds__(256) void fold_w512(
    const float* __restrict__ w5a, const float* __restrict__ g5a, const float* __restrict__ v5a,
    const float* __restrict__ w5c, const float* __restrict__ g5c, const float* __restrict__ v5c,
    u16* __restrict__ wA) {
  int idx = blockIdx.x * 256 + threadIdx.x;
  if (idx >= 2 * 32 * 4608) return;
  int br = idx / (32 * 4608);
  int r  = idx - br * (32 * 4608);
  int o  = r / 4608;
  int kk = r - o * 4608;
  int tap = kk >> 9;
  int c   = kk & 511;
  const float* w = br ? w5c : w5a;
  const float* g = br ? g5c : g5a;
  const float* v = br ? v5c : v5a;
  float scale = g[o] * rsqrtf(v[o] + 1e-3f);
  wA[idx] = f2bf(w[((size_t)o * 512 + c) * 9 + tap] * scale);
}

// fold BN scale into 32->32 conv weights; layout [br][o][tap*32 + c]
__global__ __launch_bounds__(256) void fold_w32(
    const float* __restrict__ w51, const float* __restrict__ g51, const float* __restrict__ v51,
    const float* __restrict__ w52, const float* __restrict__ g52, const float* __restrict__ v52,
    u16* __restrict__ wB) {
  int idx = blockIdx.x * 256 + threadIdx.x;
  if (idx >= 2 * 32 * 288) return;
  int br = idx / (32 * 288);
  int r  = idx - br * (32 * 288);
  int o  = r / 288;
  int kk = r - o * 288;
  int tap = kk >> 5;
  int c   = kk & 31;
  const float* w = br ? w52 : w51;
  const float* g = br ? g52 : g51;
  const float* v = br ? v52 : v51;
  float scale = g[o] * rsqrtf(v[o] + 1e-3f);
  wB[idx] = f2bf(w[((size_t)o * 32 + c) * 9 + tap] * scale);
}

__global__ __launch_bounds__(128) void fold_bias(
    const float* g5a, const float* b5a, const float* m5a, const float* v5a,
    const float* g5c, const float* b5c, const float* m5c, const float* v5c,
    const float* g51, const float* b51, const float* m51, const float* v51,
    const float* g52, const float* b52, const float* m52, const float* v52,
    float* biasA, float* biasB) {
  int t = threadIdx.x;
  if (t >= 128) return;
  int which = t >> 6, br = (t >> 5) & 1, o = t & 31;
  const float *g, *bb, *m, *v;
  float* dst;
  if (which == 0) {
    if (!br) { g = g5a; bb = b5a; m = m5a; v = v5a; } else { g = g5c; bb = b5c; m = m5c; v = v5c; }
    dst = biasA;
  } else {
    if (!br) { g = g51; bb = b51; m = m51; v = v51; } else { g = g52; bb = b52; m = m52; v = v52; }
    dst = biasB;
  }
  float scale = g[o] * rsqrtf(v[o] + 1e-3f);
  dst[br * 32 + o] = bb[o] - m[o] * scale;
}

// x f32 NCHW [4,512,64,64] -> bf16 padded NHWC [4,66,66,512] via LDS transpose
__global__ __launch_bounds__(256) void pad_x(const float* __restrict__ x, u16* __restrict__ xpad) {
  __shared__ u16 tile[64 * 65];
  int h = blockIdx.x, b = blockIdx.y, tid = threadIdx.x;
  for (int cc = 0; cc < 512; cc += 64) {
    for (int i = tid; i < 64 * 64; i += 256) {
      int cl = i >> 6, w = i & 63;
      tile[cl * 65 + w] = f2bf(x[(((size_t)b * 512 + cc + cl) * 64 + h) * 64 + w]);
    }
    __syncthreads();
    for (int i = tid; i < 64 * 64; i += 256) {
      int w = i >> 6, cl = i & 63;
      xpad[(((size_t)b * 66 + h + 1) * 66 + (w + 1)) * 512 + cc + cl] = tile[cl * 65 + w];
    }
    __syncthreads();
  }
}

// implicit-GEMM 3x3 conv 512->32, bf16 WMMA; wave computes 32x32 output tile
__global__ __launch_bounds__(64) void conv512(
    const u16* __restrict__ xpad, const u16* __restrict__ wA, const float* __restrict__ biasA,
    float* __restrict__ feat, u16* __restrict__ featbf) {
  const int lane = threadIdx.x & 31, wv = threadIdx.x >> 5;
  const int br = blockIdx.z, b = blockIdx.y;
  const int n0 = (blockIdx.x * 2 + wv) * 32;
  const int lw = lane & 15;
  const int kh = (lane >> 4) * 8;

  v8f acc[2][2];
#pragma unroll
  for (int i = 0; i < 2; ++i)
#pragma unroll
    for (int j = 0; j < 2; ++j) acc[i][j] = v8zero();

  const u16* wrow0 = wA + ((size_t)br * 32 + lw) * 4608 + kh;
  const u16* wrow1 = wrow0 + (size_t)16 * 4608;

  int hh[2], ww[2];
#pragma unroll
  for (int nt = 0; nt < 2; ++nt) { int p = n0 + nt * 16 + lw; hh[nt] = p >> 6; ww[nt] = p & 63; }

  for (int tap = 0; tap < 9; ++tap) {
    int dh = tap / 3, dw = tap - dh * 3;
    const u16* wt0 = wrow0 + tap * 512;
    const u16* wt1 = wrow1 + tap * 512;
    const u16* s0 = xpad + (((size_t)b * 66 + hh[0] + dh) * 66 + (ww[0] + dw)) * 512 + kh;
    const u16* s1 = xpad + (((size_t)b * 66 + hh[1] + dh) * 66 + (ww[1] + dw)) * 512 + kh;
#pragma unroll 4
    for (int c0 = 0; c0 < 512; c0 += 32) {
      Frag a0 = load_frag(wt0 + c0);
      Frag a1 = load_frag(wt1 + c0);
      Frag b0 = load_frag(s0 + c0);
      Frag b1 = load_frag(s1 + c0);
      acc[0][0] = WMMA_BF16(a0.v, b0.v, acc[0][0]);
      acc[0][1] = WMMA_BF16(a0.v, b1.v, acc[0][1]);
      acc[1][0] = WMMA_BF16(a1.v, b0.v, acc[1][0]);
      acc[1][1] = WMMA_BF16(a1.v, b1.v, acc[1][1]);
    }
  }
  const int rowadd = (lane >> 4) * 8;
#pragma unroll
  for (int mt = 0; mt < 2; ++mt)
#pragma unroll
    for (int nt = 0; nt < 2; ++nt) {
      int n = n0 + nt * 16 + lw;
#pragma unroll
      for (int i = 0; i < 8; ++i) {
        int c = mt * 16 + rowadd + i;
        float vl = acc[mt][nt][i] + biasA[br * 32 + c];
        vl = fmaxf(vl, 0.f);
        size_t o = ((size_t)(br * 4 + b) * 32 + c) * 4096 + n;
        feat[o] = vl;
        featbf[o] = f2bf(vl);
      }
    }
}

// q/k/v 1x1 convs on feat1 (branch 0)
__global__ __launch_bounds__(256) void qkv(
    const float* __restrict__ feat,
    const float* __restrict__ qw, const float* __restrict__ qb,
    const float* __restrict__ kw, const float* __restrict__ kb,
    const float* __restrict__ vw, const float* __restrict__ vb,
    float* __restrict__ q, float* __restrict__ k, u16* __restrict__ vbf) {
  int idx = blockIdx.x * 256 + threadIdx.x;
  if (idx >= 4 * 40 * 4096) return;
  int n = idx & 4095;
  int r = idx >> 12;
  int oc = r % 40;
  int b  = r / 40;
  const float* f = feat + ((size_t)b * 32) * 4096 + n;
  if (oc < 4) {
    float a = qb[oc];
#pragma unroll
    for (int c = 0; c < 32; ++c) a += qw[oc * 32 + c] * f[(size_t)c * 4096];
    q[((size_t)b * 4 + oc) * 4096 + n] = a;
  } else if (oc < 8) {
    int o = oc - 4;
    float a = kb[o];
#pragma unroll
    for (int c = 0; c < 32; ++c) a += kw[o * 32 + c] * f[(size_t)c * 4096];
    k[((size_t)b * 4 + o) * 4096 + n] = a;
  } else {
    int o = oc - 8;
    float a = vb[o];
#pragma unroll
    for (int c = 0; c < 32; ++c) a += vw[o * 32 + c] * f[(size_t)c * 4096];
    vbf[((size_t)b * 32 + o) * 4096 + n] = f2bf(a);
  }
}

// one attention row: energy + softmax -> bf16 att[n][m], per batch
__global__ __launch_bounds__(256) void pam_softmax(
    const float* __restrict__ q, const float* __restrict__ k, u16* __restrict__ att, int b) {
  __shared__ float e[4096];
  __shared__ float red[8];
  int n = blockIdx.x, tid = threadIdx.x;
  int lane = tid & 31, wv = tid >> 5;
  float q0 = q[((size_t)b * 4 + 0) * 4096 + n];
  float q1 = q[((size_t)b * 4 + 1) * 4096 + n];
  float q2 = q[((size_t)b * 4 + 2) * 4096 + n];
  float q3 = q[((size_t)b * 4 + 3) * 4096 + n];
  const float* k0 = k + ((size_t)b * 4) * 4096;
  const float* k1 = k0 + 4096;
  const float* k2 = k0 + 8192;
  const float* k3 = k0 + 12288;
  float mx = -3.0e38f;
  for (int m = tid; m < 4096; m += 256) {
    float s = q0 * k0[m] + q1 * k1[m] + q2 * k2[m] + q3 * k3[m];
    e[m] = s;
    mx = fmaxf(mx, s);
  }
#pragma unroll
  for (int o = 16; o > 0; o >>= 1) mx = fmaxf(mx, __shfl_xor(mx, o, 32));
  if (lane == 0) red[wv] = mx;
  __syncthreads();
  mx = red[0];
#pragma unroll
  for (int i = 1; i < 8; ++i) mx = fmaxf(mx, red[i]);
  __syncthreads();
  float sum = 0.f;
  for (int m = tid; m < 4096; m += 256) { float t = __expf(e[m] - mx); e[m] = t; sum += t; }
#pragma unroll
  for (int o = 16; o > 0; o >>= 1) sum += __shfl_xor(sum, o, 32);
  if (lane == 0) red[wv] = sum;
  __syncthreads();
  sum = red[0];
#pragma unroll
  for (int i = 1; i < 8; ++i) sum += red[i];
  float inv = 1.f / sum;
  u16* arow = att + (size_t)n * 4096;
  for (int m = tid; m < 4096; m += 256) arow[m] = f2bf(e[m] * inv);
}

// sa[c,n] = gamma * sum_m v[c,m]*att[n,m] + feat1[c,n]   (bf16 WMMA GEMM, per batch)
__global__ __launch_bounds__(64) void pam_out(
    const u16* __restrict__ vbf, const u16* __restrict__ att, const float* __restrict__ feat,
    const float* __restrict__ gamma, float* __restrict__ sa, int b) {
  const int lane = threadIdx.x & 31, wv = threadIdx.x >> 5;
  const int n0 = (blockIdx.x * 2 + wv) * 32;
  const int lw = lane & 15;
  const int kh = (lane >> 4) * 8;

  v8f acc[2][2];
#pragma unroll
  for (int i = 0; i < 2; ++i)
#pragma unroll
    for (int j = 0; j < 2; ++j) acc[i][j] = v8zero();

  const u16* a0r = vbf + ((size_t)b * 32 + lw) * 4096 + kh;
  const u16* a1r = a0r + (size_t)16 * 4096;
  const u16* b0r = att + (size_t)(n0 + lw) * 4096 + kh;
  const u16* b1r = b0r + (size_t)16 * 4096;

  for (int k0 = 0; k0 < 4096; k0 += 32) {
    Frag a0 = load_frag(a0r + k0);
    Frag a1 = load_frag(a1r + k0);
    Frag b0 = load_frag(b0r + k0);
    Frag b1 = load_frag(b1r + k0);
    acc[0][0] = WMMA_BF16(a0.v, b0.v, acc[0][0]);
    acc[0][1] = WMMA_BF16(a0.v, b1.v, acc[0][1]);
    acc[1][0] = WMMA_BF16(a1.v, b0.v, acc[1][0]);
    acc[1][1] = WMMA_BF16(a1.v, b1.v, acc[1][1]);
  }
  const int rowadd = (lane >> 4) * 8;
  float gm = gamma[0];
#pragma unroll
  for (int mt = 0; mt < 2; ++mt)
#pragma unroll
    for (int nt = 0; nt < 2; ++nt) {
      int n = n0 + nt * 16 + lw;
#pragma unroll
      for (int i = 0; i < 8; ++i) {
        int c = mt * 16 + rowadd + i;
        size_t o = ((size_t)b * 32 + c) * 4096 + n;
        sa[o] = gm * acc[mt][nt][i] + feat[o];
      }
    }
}

// channel attention on feat2 (branch 1); one workgroup per batch
__global__ __launch_bounds__(1024) void cam(
    const float* __restrict__ feat, const float* __restrict__ gamma, float* __restrict__ sc) {
  __shared__ float tile[32 * 257];
  __shared__ float energy[32 * 32];
  __shared__ float att[32 * 32];
  int b = blockIdx.x, tid = threadIdx.x;
  const float* f = feat + ((size_t)(4 + b) * 32) * 4096;
  int c = tid >> 5, d = tid & 31;
  float acc = 0.f;
  for (int n0 = 0; n0 < 4096; n0 += 256) {
    for (int i = tid; i < 32 * 256; i += 1024) {
      int cc = i >> 8, nn = i & 255;
      tile[cc * 257 + nn] = f[(size_t)cc * 4096 + n0 + nn];
    }
    __syncthreads();
    const float* tc = tile + c * 257;
    const float* td = tile + d * 257;
#pragma unroll 8
    for (int nn = 0; nn < 256; ++nn) acc += tc[nn] * td[nn];
    __syncthreads();
  }
  energy[tid] = acc;
  __syncthreads();
  if (tid < 32) {
    float emin = 3.0e38f;
    for (int j = 0; j < 32; ++j) emin = fminf(emin, energy[tid * 32 + j]);
    float s = 0.f;
    for (int j = 0; j < 32; ++j) {
      float t = __expf(emin - energy[tid * 32 + j]);  // == exp(enew - max(enew))
      att[tid * 32 + j] = t;
      s += t;
    }
    float inv = 1.f / s;
    for (int j = 0; j < 32; ++j) att[tid * 32 + j] *= inv;
  }
  __syncthreads();
  float gm = gamma[0];
  int nl = tid & 31, cr = tid >> 5;
  for (int n = nl; n < 4096; n += 32) {
    float o = 0.f;
#pragma unroll
    for (int j = 0; j < 32; ++j) o += att[cr * 32 + j] * f[(size_t)j * 4096 + n];
    sc[((size_t)b * 32 + cr) * 4096 + n] = gm * o + f[(size_t)cr * 4096 + n];
  }
}

// f32 [4][32][4096] c-major -> bf16 padded NHWC [4][66][66][32]
__global__ __launch_bounds__(64) void pad32(const float* __restrict__ src, u16* __restrict__ dst) {
  int h = blockIdx.x, b = blockIdx.y, w = threadIdx.x;
  const float* s = src + ((size_t)b * 32) * 4096 + h * 64 + w;
  u16 tmp[32];
#pragma unroll
  for (int c = 0; c < 32; ++c) tmp[c] = f2bf(s[(size_t)c * 4096]);
  u32* dp = (u32*)(dst + (((size_t)b * 66 + h + 1) * 66 + (w + 1)) * 32);
#pragma unroll
  for (int j = 0; j < 16; ++j) dp[j] = (u32)tmp[2 * j] | ((u32)tmp[2 * j + 1] << 16);
}

// implicit-GEMM 3x3 conv 32->32, bf16 WMMA
__global__ __launch_bounds__(64) void conv32(
    const u16* __restrict__ xp, const u16* __restrict__ wB, const float* __restrict__ biasB,
    float* __restrict__ outv) {
  const int lane = threadIdx.x & 31, wv = threadIdx.x >> 5;
  const int br = blockIdx.z, b = blockIdx.y;
  const int n0 = (blockIdx.x * 2 + wv) * 32;
  const int lw = lane & 15;
  const int kh = (lane >> 4) * 8;

  v8f acc[2][2];
#pragma unroll
  for (int i = 0; i < 2; ++i)
#pragma unroll
    for (int j = 0; j < 2; ++j) acc[i][j] = v8zero();

  const u16* wrow0 = wB + ((size_t)br * 32 + lw) * 288 + kh;
  const u16* wrow1 = wrow0 + (size_t)16 * 288;
  const u16* xbase = xp + ((size_t)br * 4 + b) * 66 * 66 * 32;

  int hh[2], ww[2];
#pragma unroll
  for (int nt = 0; nt < 2; ++nt) { int p = n0 + nt * 16 + lw; hh[nt] = p >> 6; ww[nt] = p & 63; }

#pragma unroll
  for (int tap = 0; tap < 9; ++tap) {
    int dh = tap / 3, dw = tap - dh * 3;
    Frag a0 = load_frag(wrow0 + tap * 32);
    Frag a1 = load_frag(wrow1 + tap * 32);
    Frag b0 = load_frag(xbase + ((size_t)(hh[0] + dh) * 66 + (ww[0] + dw)) * 32 + kh);
    Frag b1 = load_frag(xbase + ((size_t)(hh[1] + dh) * 66 + (ww[1] + dw)) * 32 + kh);
    acc[0][0] = WMMA_BF16(a0.v, b0.v, acc[0][0]);
    acc[0][1] = WMMA_BF16(a0.v, b1.v, acc[0][1]);
    acc[1][0] = WMMA_BF16(a1.v, b0.v, acc[1][0]);
    acc[1][1] = WMMA_BF16(a1.v, b1.v, acc[1][1]);
  }
  const int rowadd = (lane >> 4) * 8;
#pragma unroll
  for (int mt = 0; mt < 2; ++mt)
#pragma unroll
    for (int nt = 0; nt < 2; ++nt) {
      int n = n0 + nt * 16 + lw;
#pragma unroll
      for (int i = 0; i < 8; ++i) {
        int c = mt * 16 + rowadd + i;
        float vl = fmaxf(acc[mt][nt][i] + biasB[br * 32 + c], 0.f);
        outv[((size_t)(br * 4 + b) * 32 + c) * 4096 + n] = vl;
      }
    }
}

// out[b,o,n] = relu(b8[o] + sum_c w8[o,c]*(sa_conv+sc_conv)[b,c,n])
__global__ __launch_bounds__(256) void final1x1(
    const float* __restrict__ convo, const float* __restrict__ w8, const float* __restrict__ b8,
    float* __restrict__ out) {
  __shared__ float sw[64 * 32];
  __shared__ float sb[64];
  int tid = threadIdx.x;
  for (int i = tid; i < 2048; i += 256) sw[i] = w8[i];
  if (tid < 64) sb[tid] = b8[tid];
  __syncthreads();
  int idx = blockIdx.x * 256 + tid;  // 4*4096
  int n = idx & 4095;
  int b = idx >> 12;
  float fs[32];
#pragma unroll
  for (int c = 0; c < 32; ++c)
    fs[c] = convo[((size_t)b * 32 + c) * 4096 + n] + convo[((size_t)(4 + b) * 32 + c) * 4096 + n];
#pragma unroll 4
  for (int o = 0; o < 64; ++o) {
    float a = sb[o];
#pragma unroll
    for (int c = 0; c < 32; ++c) a += sw[o * 32 + c] * fs[c];
    out[((size_t)b * 64 + o) * 4096 + n] = fmaxf(a, 0.f);
  }
}

// ---------------- launch ----------------
extern "C" void kernel_launch(void* const* d_in, const int* in_sizes, int n_in,
                              void* d_out, int out_size, void* d_ws, size_t ws_size,
                              hipStream_t stream) {
  const float* x    = (const float*)d_in[0];
  const float* w5a  = (const float*)d_in[1];
  const float* g5a  = (const float*)d_in[2];
  const float* b5a  = (const float*)d_in[3];
  const float* m5a  = (const float*)d_in[4];
  const float* v5a  = (const float*)d_in[5];
  const float* w5c  = (const float*)d_in[6];
  const float* g5c  = (const float*)d_in[7];
  const float* b5c  = (const float*)d_in[8];
  const float* m5c  = (const float*)d_in[9];
  const float* v5c  = (const float*)d_in[10];
  const float* qw   = (const float*)d_in[11];
  const float* qb   = (const float*)d_in[12];
  const float* kw   = (const float*)d_in[13];
  const float* kb   = (const float*)d_in[14];
  const float* vw   = (const float*)d_in[15];
  const float* vb   = (const float*)d_in[16];
  const float* gpam = (const float*)d_in[17];
  const float* gcam = (const float*)d_in[18];
  const float* w51  = (const float*)d_in[19];
  const float* g51  = (const float*)d_in[20];
  const float* b51  = (const float*)d_in[21];
  const float* m51  = (const float*)d_in[22];
  const float* v51  = (const float*)d_in[23];
  const float* w52  = (const float*)d_in[24];
  const float* g52  = (const float*)d_in[25];
  const float* b52  = (const float*)d_in[26];
  const float* m52  = (const float*)d_in[27];
  const float* v52  = (const float*)d_in[28];
  const float* w8   = (const float*)d_in[29];
  const float* b8   = (const float*)d_in[30];

  char* ws = (char*)d_ws;
  u16*   xpad   = (u16*)(ws + OFF_XPAD);
  u16*   p32    = (u16*)(ws + OFF_PAD32);
  u16*   wA     = (u16*)(ws + OFF_WA);
  u16*   wB     = (u16*)(ws + OFF_WB);
  float* biasA  = (float*)(ws + OFF_BIASA);
  float* biasB  = (float*)(ws + OFF_BIASB);
  float* feat   = (float*)(ws + OFF_FEAT);
  u16*   featbf = (u16*)(ws + OFF_FEATBF);
  float* qv     = (float*)(ws + OFF_Q);
  float* kv     = (float*)(ws + OFF_K);
  u16*   vbf    = (u16*)(ws + OFF_VBF);
  u16*   att    = (u16*)(ws + OFF_ATT);
  float* sa     = (float*)(ws + OFF_SA);
  float* sc     = (float*)(ws + OFF_SC);
  float* convo  = (float*)(ws + OFF_CONVO);
  float* out    = (float*)d_out;

  // zero padded-input regions (xpad + p32 are adjacent)
  long n16 = (long)((SZ_XPAD + SZ_PAD32) / 16);
  zero_u4<<<(n16 + 255) / 256, 256, 0, stream>>>((U4*)(ws + OFF_XPAD), n16);

  fold_w512<<<(2 * 32 * 4608 + 255) / 256, 256, 0, stream>>>(w5a, g5a, v5a, w5c, g5c, v5c, wA);
  fold_w32<<<(2 * 32 * 288 + 255) / 256, 256, 0, stream>>>(w51, g51, v51, w52, g52, v52, wB);
  fold_bias<<<1, 128, 0, stream>>>(g5a, b5a, m5a, v5a, g5c, b5c, m5c, v5c,
                                   g51, b51, m51, v51, g52, b52, m52, v52, biasA, biasB);

  pad_x<<<dim3(64, 4), 256, 0, stream>>>(x, xpad);

  conv512<<<dim3(64, 4, 2), 64, 0, stream>>>(xpad, wA, biasA, feat, featbf);

  qkv<<<(4 * 40 * 4096 + 255) / 256, 256, 0, stream>>>(feat, qw, qb, kw, kb, vw, vb, qv, kv, vbf);

  cam<<<4, 1024, 0, stream>>>(feat, gcam, sc);

  for (int b = 0; b < 4; ++b) {
    pam_softmax<<<4096, 256, 0, stream>>>(qv, kv, att, b);
    pam_out<<<64, 64, 0, stream>>>(vbf, att, feat, gpam, sa, b);
  }

  pad32<<<dim3(64, 4), 64, 0, stream>>>(sa, p32);
  pad32<<<dim3(64, 4), 64, 0, stream>>>(sc, p32 + (size_t)4 * 66 * 66 * 32);

  conv32<<<dim3(64, 4, 2), 64, 0, stream>>>(p32, wB, biasB, convo);

  final1x1<<<64, 256, 0, stream>>>(convo, w8, b8, out);

  (void)in_sizes; (void)n_in; (void)out_size; (void)ws_size;
}
